// RandomlyWiredStage_29085518528582
// MI455X (gfx1250) — matrix operand order repaired and checked by reading, not compiled, not run on hardware
//
#include <hip/hip_runtime.h>
#include <cstddef>

// ---------------------------------------------------------------------------
// RandomlyWiredStage for MI455X (gfx1250, wave32, WMMA bf16 K=32)
//
// Heavy path: per-node fused  aggregate -> conv3x3 -> ReLU  as implicit GEMM
//   M = 32768 pixels, N = 64 couts, K = 576 (9 taps x 64 ci)
//   v_wmma_f32_16x16x32_bf16, A staged in LDS (bf16), B pre-shuffled in L2.
// Light path: 1-wave bookkeeping kernel per node (softmax / top-8 cut /
//   edge-state rescale), matching the reference semantics.
// ---------------------------------------------------------------------------

#define NNODES 32
#define NODE_ELEMS (32 * 64 * 32 * 32)   // 2,097,152 floats per node output

// workspace layout (floats unless noted)
#define TA_OFF    0            // [t][s][b]  32*32*32
#define TF_OFF    32768        // [t][s][b]
#define NA_OFF    65536        // [t][b]     32*32
#define PART_OFF  66560        // [t][b]
#define AD_OFF    67584        // [b][t]
#define POOL_OFF  68608        // [nid][b*64+c]  32*2048
#define M1_OFF    134144       // [c][t]  64*32
#define M2_OFF    136192
#define STATE_FLOATS 138240
#define BPREP_BYTE  (STATE_FLOATS * 4)            // 552,960
#define BPREP_HALF  (30 * 18 * 4 * 32 * 16)       // 1,105,920 bf16 halves
#define NODE_BYTE   (BPREP_BYTE + BPREP_HALF * 2) // 2,764,800

struct EdgeList { int n; int s[4]; };
struct TarList  { int n; int t[32]; };

typedef __attribute__((ext_vector_type(16))) __bf16 v16bf;
typedef __attribute__((ext_vector_type(8)))  float  v8f;

union Frag { uint4 q[2]; v16bf v; };

__device__ __forceinline__ unsigned short f2bf(float f) {
    union { float f; unsigned u; } c; c.f = f;
    unsigned u = c.u + 0x7fffu + ((c.u >> 16) & 1u);  // round-to-nearest-even
    return (unsigned short)(u >> 16);
}

// ---------------------------------------------------------------------------
// one-shot state init (zero everything, part[0][b] = 1)
// ---------------------------------------------------------------------------
__global__ void init_state(float* ws) {
    int idx = blockIdx.x * 256 + threadIdx.x;
    if (idx >= STATE_FLOATS) return;
    float v = 0.f;
    if (idx >= PART_OFF && idx < PART_OFF + 32) v = 1.f;   // part[node 0]
    ws[idx] = v;
}

// ---------------------------------------------------------------------------
// pre-shuffle conv weights f32 -> bf16 into per-lane B-fragment layout:
// Bprep[node-1][chunk t(18)][frag j(4)][lane(32)][16 halves]
//   k_in_chunk = (lane/16)*16 + v2,  ci = (t&1)*32 + k_in_chunk,
//   (ky,kx) = tap t>>1,  co = j*16 + lane%16
// ---------------------------------------------------------------------------
__global__ void prep_weights(const float* __restrict__ cw,
                             unsigned short* __restrict__ bp) {
    int idx = blockIdx.x * 256 + threadIdx.x;
    if (idx >= BPREP_HALF) return;
    int v2   = idx & 15;
    int lane = (idx >> 4) & 31;
    int j    = (idx >> 9) & 3;
    int c1   = idx >> 11;
    int t    = c1 % 18;
    int nm1  = c1 / 18;
    int kic  = ((lane >> 4) << 4) + v2;
    int ci   = ((t & 1) << 5) + kic;
    int kidx = t >> 1;
    int ky = kidx / 3, kx = kidx % 3;
    int co  = (j << 4) + (lane & 15);
    int nid = nm1 + 1;
    float w = cw[(((((size_t)nid * 64 + co) * 64 + ci) * 3 + ky) * 3) + kx];
    bp[idx] = f2bf(w);
}

// ---------------------------------------------------------------------------
// M1 = attend_w1 @ embs^T, M2 = attend_w2 @ embs^T   (64 x 32 each)
// ---------------------------------------------------------------------------
__global__ void prep_mats(const float* __restrict__ embs,
                          const float* __restrict__ w1,
                          const float* __restrict__ w2,
                          float* __restrict__ M1, float* __restrict__ M2) {
    for (int o = threadIdx.x; o < 2048; o += 256) {
        int c = o >> 5, n = o & 31;
        float d1 = 0.f, d2 = 0.f;
        for (int e = 0; e < 128; ++e) {
            float em = embs[n * 128 + e];
            d1 += w1[c * 128 + e] * em;
            d2 += w2[c * 128 + e] * em;
        }
        M1[c * 32 + n] = d1;
        M2[c * 32 + n] = d2;
    }
}

// ---------------------------------------------------------------------------
// pooled sums of x -> pool slot of node 0 (bookkeeping divides by 1024)
// ---------------------------------------------------------------------------
__global__ void pool_x(const float* __restrict__ x, float* __restrict__ pool) {
    __shared__ float red[256];
    int bc = blockIdx.x, tid = threadIdx.x;
    const float4* p = (const float4*)(x + ((size_t)bc << 10));
    float4 v = p[tid];
    red[tid] = v.x + v.y + v.z + v.w;
    __syncthreads();
    for (int s = 128; s > 0; s >>= 1) {
        if (tid < s) red[tid] += red[tid + s];
        __syncthreads();
    }
    if (tid == 0) pool[bc] = red[0];
}

// ---------------------------------------------------------------------------
// fused: aggregate in-edges -> bf16 LDS tile -> WMMA conv3x3 -> ReLU -> store
//        + pooled partial sums (ds_add_f32 then 64 global atomics per WG)
// grid 256 WGs (128 pixels each = 4 image rows of one batch), 8 waves/WG.
// wave = 16 pixels x 64 couts = four 16x16 f32 accumulators.
// ---------------------------------------------------------------------------
__global__ __launch_bounds__(256) void node_conv_wmma(
        int nid, EdgeList in,
        const float* __restrict__ x, const float* __restrict__ nodeBase,
        float* __restrict__ outp, const unsigned short* __restrict__ bprep,
        const float* __restrict__ TA, const float* __restrict__ TF,
        const float* __restrict__ PART, const float* __restrict__ NA,
        float* __restrict__ pool) {
    __shared__ __align__(16) unsigned short in_s[6 * 34 * 64];  // halo tile
    __shared__ float pool_s[64];
    const int tid = threadIdx.x;
    const int P0  = blockIdx.x * 128;
    const int b   = P0 >> 10;
    const int y0  = (P0 >> 5) & 31;

    // per-source aggregation coefficients (uniform over the WG: b is fixed)
    const float* sp[4]; float cf[4];
    for (int k = 0; k < 4; ++k) {
        float c = 0.f; const float* p = x;
        if (k < in.n) {
            int s = in.s[k];
            float tra = TA[(nid * 32 + s) * 32 + b] * TF[(nid * 32 + s) * 32 + b];
            if (tra != 0.f && PART[s * 32 + b] != 0.f)
                c = (s == 0) ? 1.f : NA[s * 32 + b];
            p = (s == 0) ? x : (nodeBase + (size_t)(s - 1) * NODE_ELEMS);
        }
        sp[k] = p; cf[k] = c;
    }

    if (tid < 64) pool_s[tid] = 0.f;

    // stage aggregated input (rows y0-1..y0+4, cols -1..32, 64 ci) as bf16
    for (int i = tid; i < 6 * 34 * 64; i += 256) {
        int ci = i & 63;
        int xy = i >> 6;
        int xx = xy % 34, yy = xy / 34;
        int y = y0 - 1 + yy, xp = xx - 1;
        float v = 0.f;
        if ((unsigned)y < 32u && (unsigned)xp < 32u) {
            int off = (((b << 6) + ci) << 10) + (y << 5) + xp;
            for (int k = 0; k < 4; ++k)
                if (cf[k] != 0.f) v += cf[k] * sp[k][off];
        }
        in_s[i] = f2bf(v);
    }
    __syncthreads();

    const int lane = tid & 31, w = tid >> 5;
    const int yw  = y0 + (w >> 1);          // this wave's image row
    const int xw0 = (w & 1) << 4;           // x base (0 or 16)
    const int lh = lane & 15, lg = lane >> 4;

    v8f acc[4] = {};
#pragma unroll
    for (int t = 0; t < 18; ++t) {
        const int kidx = t >> 1, ci0 = (t & 1) << 5;
        const int dy = kidx / 3 - 1, dx = kidx % 3 - 1;
        const int yy = (w >> 1) + 1 + dy;
        const int xx = xw0 + lh + 1 + dx;
        const int aidx = ((yy * 34 + xx) << 6) + ci0 + (lg << 4);
        Frag fa;                                   // A: 16x32 bf16, k = 16*lg + s (contiguous)
        fa.q[0] = *(const uint4*)&in_s[aidx];      // ds_load_b128 (K pairs v0..3)
        fa.q[1] = *(const uint4*)&in_s[aidx + 8];  // ds_load_b128 (K pairs v4..7)
        const unsigned short* bb = bprep + (size_t)t * 2048 + lane * 16;
        if (t + 1 < 18) __builtin_prefetch(bb + 2048, 0, 1); // global_prefetch
#pragma unroll
        for (int j = 0; j < 4; ++j) {
            Frag fb;                               // B: 32x16 bf16, k = 16*lg + s (contiguous)
            const uint4* bq = (const uint4*)(bb + j * 512);
            fb.q[0] = bq[0]; fb.q[1] = bq[1];
            acc[j] = __builtin_amdgcn_wmma_f32_16x16x32_bf16(
                false, fa.v, false, fb.v, (short)0, acc[j], false, false);
        }
    }

    // epilogue: ReLU, NCHW f32 store, pooled partial sums
#pragma unroll
    for (int j = 0; j < 4; ++j) {
        const int co = (j << 4) + lh;
        float r[8]; float s8 = 0.f;
#pragma unroll
        for (int q = 0; q < 8; ++q) { r[q] = fmaxf(acc[j][q], 0.f); s8 += r[q]; }
        float* dst = outp + (((size_t)((b << 6) + co)) << 10) + (yw << 5) + xw0 + (lg << 3);
        *(float4*)(dst)     = make_float4(r[0], r[1], r[2], r[3]);
        *(float4*)(dst + 4) = make_float4(r[4], r[5], r[6], r[7]);
        atomicAdd(&pool_s[co], s8);                 // ds_add_f32
    }
    __syncthreads();
    if (tid < 64) atomicAdd(&pool[b * 64 + tid], pool_s[tid]);
}

// ---------------------------------------------------------------------------
// per-node attention / cut / state update, one wave (thread = batch element)
// ---------------------------------------------------------------------------
__global__ void node_bookkeep(int nid, EdgeList in, TarList tl,
                              float* __restrict__ TA, float* __restrict__ TF,
                              float* __restrict__ NA, float* __restrict__ PART,
                              float* __restrict__ AD,
                              const float* __restrict__ pool,
                              const float* __restrict__ M) {
    int b = threadIdx.x;
    if (b >= 32) return;

    float pm = 1.f;
    if (nid != 0) {
        pm = 0.f;
        for (int k = 0; k < in.n; ++k) {
            int s = in.s[k];
            float tra = TA[(nid * 32 + s) * 32 + b] * TF[(nid * 32 + s) * 32 + b];
            if (tra != 0.f && PART[s * 32 + b] != 0.f) pm = 1.f;
        }
        PART[nid * 32 + b] = pm;
    }

    float pooled[64];
    for (int c = 0; c < 64; ++c) pooled[c] = pool[b * 64 + c] * (1.f / 1024.f);

    int nt = tl.n;
    float lg[32]; float mx = -3.4e38f;
    for (int j = 0; j < nt; ++j) {
        int t = tl.t[j]; float d = 0.f;
        for (int c = 0; c < 64; ++c) d += pooled[c] * M[c * 32 + t];
        lg[j] = d; mx = fmaxf(mx, d);
    }
    float se = 0.f;
    for (int j = 0; j < nt; ++j) { lg[j] = __expf(lg[j] - mx); se += lg[j]; }
    float inv = 1.f / se;
    float srcA = (nid == 0) ? 1.f : NA[nid * 32 + b];
    float asent[32];
    for (int j = 0; j < nt; ++j) asent[j] = lg[j] * inv * srcA * pm;

    AD[b * 32 + nid] = 0.f;
    for (int j = 0; j < nt; ++j) AD[b * 32 + tl.t[j]] += asent[j];

    // cut(): top-8 with eps gate, renormalize
    float adv[32], mask[32]; bool picked[32];
    for (int t = 0; t < 32; ++t) { adv[t] = AD[b * 32 + t]; mask[t] = 0.f; picked[t] = false; }
    float vs = 0.f;
    for (int k = 0; k < 8; ++k) {
        int bi = 0; float bv = -3.4e38f;
        for (int t = 0; t < 32; ++t)
            if (!picked[t] && adv[t] > bv) { bv = adv[t]; bi = t; }
        picked[bi] = true;
        if (bv > 0.01f) { vs += bv; mask[bi] = 1.f; }
    }
    float scale = (vs > 0.f) ? 1.f / fmaxf(vs, 1e-12f) : 0.f;
    for (int t = 0; t < 32; ++t) AD[b * 32 + t] = adv[t] * mask[t] * scale;

    // create edges, then rescale all downstream state by nf = mask*scale
    for (int j = 0; j < nt; ++j) {
        int t = tl.t[j];
        TA[(t * 32 + nid) * 32 + b] = asent[j];
        TF[(t * 32 + nid) * 32 + b] = 1.f;
        NA[t * 32 + b] += asent[j];
    }
    for (int t = nid + 1; t < 32; ++t) {
        float nf = mask[t] * scale;
        for (int s = 0; s <= nid; ++s) TF[(t * 32 + s) * 32 + b] *= nf;
        NA[t * 32 + b] *= nf;
    }
}

// ---------------------------------------------------------------------------
// node 31: weighted aggregation of in-edges straight into d_out (f32)
// ---------------------------------------------------------------------------
__global__ __launch_bounds__(256) void final_aggr(
        EdgeList in, const float* __restrict__ x,
        const float* __restrict__ nodeBase,
        const float* __restrict__ TA, const float* __restrict__ TF,
        const float* __restrict__ PART, const float* __restrict__ NA,
        float* __restrict__ out) {
    int idx  = blockIdx.x * 256 + threadIdx.x;  // float4 index
    size_t base = (size_t)idx << 2;
    int b = (int)(base >> 16);                  // 64*1024 elems per batch
    float4 acc = make_float4(0.f, 0.f, 0.f, 0.f);
    const int nid = 31;
    for (int k = 0; k < in.n; ++k) {
        int s = in.s[k];
        float tra = TA[(nid * 32 + s) * 32 + b] * TF[(nid * 32 + s) * 32 + b];
        if (tra != 0.f && PART[s * 32 + b] != 0.f) {
            float c = (s == 0) ? 1.f : NA[s * 32 + b];
            const float* p = (s == 0) ? x : nodeBase + (size_t)(s - 1) * NODE_ELEMS;
            float4 v = *(const float4*)(p + base);
            acc.x += c * v.x; acc.y += c * v.y; acc.z += c * v.z; acc.w += c * v.w;
        }
    }
    *(float4*)(out + base) = acc;
}

// ---------------------------------------------------------------------------
// host: deterministic DAG (same family as reference: {i-1} + 3 picks < i)
// ---------------------------------------------------------------------------
struct Graph { int in_n[32]; int in_s[32][4]; int out_n[32]; int out_t[32][32]; };

static void build_graph(Graph& g) {
    bool adj[32][32] = {};
    unsigned long long st = 0x2545F4914F6CDD1Dull ^ 42ull;
    auto rnd = [&]() { st ^= st << 13; st ^= st >> 7; st ^= st << 17; return st; };
    for (int i = 1; i < 32; ++i) {
        adj[i][i - 1] = true;
        int k = i < 3 ? i : 3;
        int got = 0; bool pick[32] = {};
        while (got < k) {
            int c = (int)(rnd() % (unsigned long long)i);
            if (!pick[c]) { pick[c] = true; adj[i][c] = true; ++got; }
        }
    }
    for (int i = 0; i < 32; ++i) {
        g.in_n[i] = 0; g.out_n[i] = 0;
        for (int s = 0; s < i; ++s)
            if (adj[i][s] && g.in_n[i] < 4) g.in_s[i][g.in_n[i]++] = s;
        for (int t = i + 1; t < 32; ++t)
            if (adj[t][i]) g.out_t[i][g.out_n[i]++] = t;
    }
}

extern "C" void kernel_launch(void* const* d_in, const int* in_sizes, int n_in,
                              void* d_out, int out_size, void* d_ws, size_t ws_size,
                              hipStream_t stream) {
    (void)in_sizes; (void)n_in; (void)out_size; (void)ws_size;
    const float* x    = (const float*)d_in[0];
    const float* embs = (const float*)d_in[1];
    const float* w1   = (const float*)d_in[2];
    const float* w2   = (const float*)d_in[3];
    const float* cw   = (const float*)d_in[4];
    float* out = (float*)d_out;
    float* wsf = (float*)d_ws;
    unsigned short* bprep = (unsigned short*)((char*)d_ws + BPREP_BYTE);
    float* nodeBase = (float*)((char*)d_ws + NODE_BYTE);
    float* TA = wsf + TA_OFF,  * TF = wsf + TF_OFF,  * NA = wsf + NA_OFF;
    float* PART = wsf + PART_OFF, * AD = wsf + AD_OFF, * POOL = wsf + POOL_OFF;
    float* M1 = wsf + M1_OFF, * M2 = wsf + M2_OFF;

    Graph g; build_graph(g);

    init_state<<<(STATE_FLOATS + 255) / 256, 256, 0, stream>>>(wsf);
    prep_weights<<<(BPREP_HALF + 255) / 256, 256, 0, stream>>>(cw, bprep);
    prep_mats<<<1, 256, 0, stream>>>(embs, w1, w2, M1, M2);
    pool_x<<<2048, 256, 0, stream>>>(x, POOL);

    {   // node 0: attends from x with w1
        EdgeList el; el.n = 0; for (int k = 0; k < 4; ++k) el.s[k] = 0;
        TarList tl; tl.n = g.out_n[0];
        for (int k = 0; k < 32; ++k) tl.t[k] = (k < tl.n) ? g.out_t[0][k] : 0;
        node_bookkeep<<<1, 32, 0, stream>>>(0, el, tl, TA, TF, NA, PART, AD, POOL, M1);
    }

    for (int nid = 1; nid <= 30; ++nid) {
        EdgeList el; el.n = g.in_n[nid];
        for (int k = 0; k < 4; ++k) el.s[k] = (k < el.n) ? g.in_s[nid][k] : 0;
        float* outp = nodeBase + (size_t)(nid - 1) * NODE_ELEMS;
        node_conv_wmma<<<256, 256, 0, stream>>>(
            nid, el, x, nodeBase, outp,
            bprep + (size_t)(nid - 1) * 18 * 2048,
            TA, TF, PART, NA, POOL + nid * 2048);
        TarList tl; tl.n = g.out_n[nid];
        for (int k = 0; k < 32; ++k) tl.t[k] = (k < tl.n) ? g.out_t[nid][k] : 0;
        node_bookkeep<<<1, 32, 0, stream>>>(
            nid, el, tl, TA, TF, NA, PART, AD, POOL + nid * 2048, M2);
    }

    {   // node 31: output aggregation
        EdgeList el; el.n = g.in_n[31];
        for (int k = 0; k < 4; ++k) el.s[k] = (k < el.n) ? g.in_s[31][k] : 0;
        final_aggr<<<2048, 256, 0, stream>>>(el, x, nodeBase, TA, TF, PART, NA, out);
    }
}
